// POFDConv_69758858822414
// MI455X (gfx1250) — matrix operand order repaired and checked by reading, not compiled, not run on hardware
//
#include <hip/hip_runtime.h>

#define NNODES 50000
#define NEDGES 800000
#define FIN    128
#define FOUT   64
#define NHEADS 4
#define ALPHA  0.01f

typedef __attribute__((ext_vector_type(16))) __bf16 v16bf;
typedef __attribute__((ext_vector_type(8)))  float  v8f;

// ---- monotonic float <-> uint key for atomicMax-based segment max ----
__device__ __forceinline__ unsigned fkey(float f) {
    unsigned b = __float_as_uint(f);
    return (b & 0x80000000u) ? ~b : (b | 0x80000000u);
}
__device__ __forceinline__ float funkey(unsigned k) {
    unsigned b = (k & 0x80000000u) ? (k & 0x7fffffffu) : ~k;
    return __uint_as_float(b);
}

__global__ void fill_u32_kernel(unsigned* __restrict__ p, unsigned v, int n) {
    int i = blockIdx.x * blockDim.x + threadIdx.x;
    if (i < n) p[i] = v;
}

// load one 16-bit WMMA A-fragment (16x32) for a row of A, converting f32->bf16
__device__ __forceinline__ v16bf load_a_frag(const float* __restrict__ arow, int k0) {
    const float4 f0 = *(const float4*)(arow + k0);
    const float4 f1 = *(const float4*)(arow + k0 + 4);
    const float4 f2 = *(const float4*)(arow + k0 + 16);
    const float4 f3 = *(const float4*)(arow + k0 + 20);
    v16bf a;
    a[0]=(__bf16)f0.x;  a[1]=(__bf16)f0.y;  a[2]=(__bf16)f0.z;  a[3]=(__bf16)f0.w;
    a[4]=(__bf16)f1.x;  a[5]=(__bf16)f1.y;  a[6]=(__bf16)f1.z;  a[7]=(__bf16)f1.w;
    a[8]=(__bf16)f2.x;  a[9]=(__bf16)f2.y;  a[10]=(__bf16)f2.z; a[11]=(__bf16)f2.w;
    a[12]=(__bf16)f3.x; a[13]=(__bf16)f3.y; a[14]=(__bf16)f3.z; a[15]=(__bf16)f3.w;
    return a;
}

// ---------------------------------------------------------------------------
// D(nrows x 64) = A(nrows x K, f32) * W[head](K x 64, f32) via bf16 WMMA.
// Epilogue fuses s1 = D @ attn[0:64], s2 = D @ attn[64:128].
// Block: 256 threads = 8 waves; each wave owns TWO 16-row tiles (M=32) so the
// four B fragments of each k-step are reused twice (8 WMMAs per 8 ds_load_b128).
// W staged in LDS pre-swizzled into the 16-bit B-fragment layout.
// ---------------------------------------------------------------------------
__global__ __launch_bounds__(256) void gat_gemm_wmma(
    const float* __restrict__ A, const float* __restrict__ W,
    const float* __restrict__ attn, float* __restrict__ Hout,
    float* __restrict__ s1, float* __restrict__ s2, int nrows, int K)
{
    __shared__ __bf16 Wl[16384];            // (K/32)*4*32*16 bf16, K<=256 -> 32KB
    const int head = blockIdx.y;
    const int tid  = threadIdx.x;

    // Stage W -> LDS in B-fragment order: frag(s,t): lane holds 16
    // bf16 = B[k = s*32 + (lane>=16?8:0) + (j&7) + (j>=8?16:0)][col = t*16 + lane%16]
    const float* Wh = W + (size_t)head * K * 64;
    for (int idx = tid; idx < K * 64; idx += 256) {
        int j    = idx & 15;
        int lane = (idx >> 4) & 31;
        int t    = (idx >> 9) & 3;
        int s    = idx >> 11;
        int k    = s * 32 + (lane >> 4) * 8 + (j & 7) + (j >> 3) * 16;
        int c    = t * 16 + (lane & 15);
        Wl[idx]  = (__bf16)Wh[k * 64 + c];
    }
    __syncthreads();

    const int wave = tid >> 5, lane = tid & 31;
    const int half = lane >> 4, n = lane & 15;
    const int row0 = blockIdx.x * 256 + wave * 32;

    // A-fragment rows for the two row-tiles (clamped at tail; stores guarded)
    int ar0 = row0 + n;      if (ar0 >= nrows) ar0 = nrows - 1;
    int ar1 = row0 + 16 + n; if (ar1 >= nrows) ar1 = nrows - 1;
    const float* arow0 = A + (size_t)ar0 * K;
    const float* arow1 = A + (size_t)ar1 * K;

    v8f acc[2][4];
    #pragma unroll
    for (int mt = 0; mt < 2; ++mt)
        #pragma unroll
        for (int t = 0; t < 4; ++t)
            #pragma unroll
            for (int r = 0; r < 8; ++r) acc[mt][t][r] = 0.0f;

    const int ksteps = K >> 5;
    for (int s = 0; s < ksteps; ++s) {
        // lanes 0-15 take K {0..7,16..23}; lanes 16-31 take {8..15,24..31}
        const int k0 = s * 32 + half * 8;
        const v16bf a0 = load_a_frag(arow0, k0);
        const v16bf a1 = load_a_frag(arow1, k0);
        // issue all four B-fragment loads before any WMMA (partial dscnt waits)
        const __bf16* bb = &Wl[((s * 4) * 32 + lane) << 4];
        const v16bf b0 = *(const v16bf*)(bb);
        const v16bf b1 = *(const v16bf*)(bb + 512);
        const v16bf b2 = *(const v16bf*)(bb + 1024);
        const v16bf b3 = *(const v16bf*)(bb + 1536);
        acc[0][0] = __builtin_amdgcn_wmma_f32_16x16x32_bf16(false, a0, false, b0, (short)0, acc[0][0], false, false);
        acc[1][0] = __builtin_amdgcn_wmma_f32_16x16x32_bf16(false, a1, false, b0, (short)0, acc[1][0], false, false);
        acc[0][1] = __builtin_amdgcn_wmma_f32_16x16x32_bf16(false, a0, false, b1, (short)0, acc[0][1], false, false);
        acc[1][1] = __builtin_amdgcn_wmma_f32_16x16x32_bf16(false, a1, false, b1, (short)0, acc[1][1], false, false);
        acc[0][2] = __builtin_amdgcn_wmma_f32_16x16x32_bf16(false, a0, false, b2, (short)0, acc[0][2], false, false);
        acc[1][2] = __builtin_amdgcn_wmma_f32_16x16x32_bf16(false, a1, false, b2, (short)0, acc[1][2], false, false);
        acc[0][3] = __builtin_amdgcn_wmma_f32_16x16x32_bf16(false, a0, false, b3, (short)0, acc[0][3], false, false);
        acc[1][3] = __builtin_amdgcn_wmma_f32_16x16x32_bf16(false, a1, false, b3, (short)0, acc[1][3], false, false);
    }

    // Epilogue: store h (f32) + fused attention dot products.
    // C/D layout: VGPR r, lanes 0-15 -> row r, col=lane; lanes 16-31 -> row 8+r.
    const float* at = attn + head * 128;
    float p1[2][8], p2[2][8];
    #pragma unroll
    for (int mt = 0; mt < 2; ++mt)
        #pragma unroll
        for (int r = 0; r < 8; ++r) { p1[mt][r] = 0.f; p2[mt][r] = 0.f; }

    #pragma unroll
    for (int t = 0; t < 4; ++t) {
        const int col = t * 16 + n;
        const float w1 = at[col], w2 = at[64 + col];
        #pragma unroll
        for (int mt = 0; mt < 2; ++mt) {
            #pragma unroll
            for (int r = 0; r < 8; ++r) {
                const float v = acc[mt][t][r];
                const int row = row0 + mt * 16 + half * 8 + r;
                if (row < nrows) Hout[((size_t)head * nrows + row) * 64 + col] = v;
                p1[mt][r] += v * w1;
                p2[mt][r] += v * w2;
            }
        }
    }
    // butterfly-reduce over each 16-lane half (xor masks stay within the half)
    #pragma unroll
    for (int mt = 0; mt < 2; ++mt) {
        #pragma unroll
        for (int r = 0; r < 8; ++r) {
            #pragma unroll
            for (int m = 1; m < 16; m <<= 1) {
                p1[mt][r] += __shfl_xor(p1[mt][r], m, 32);
                p2[mt][r] += __shfl_xor(p2[mt][r], m, 32);
            }
        }
    }
    if (n == 0) {
        #pragma unroll
        for (int mt = 0; mt < 2; ++mt) {
            #pragma unroll
            for (int r = 0; r < 8; ++r) {
                const int row = row0 + mt * 16 + half * 8 + r;
                if (row < nrows) {
                    s1[(size_t)head * nrows + row] = p1[mt][r];
                    s2[(size_t)head * nrows + row] = p2[mt][r];
                }
            }
        }
    }
}

// ---------------------------------------------------------------------------
// Edge passes. gridDim.y = head.
// ---------------------------------------------------------------------------
__device__ __forceinline__ float edge_score(const float* s1, const float* s2,
                                            int h, int N, int sr, int ds) {
    float s = s1[(size_t)h * N + sr] + s2[(size_t)h * N + ds];
    float lr = s > 0.f ? s : ALPHA * s;   // leaky_relu
    return -lr;                            // reference negates
}

__global__ void edge_attn_max(const int* __restrict__ src, const int* __restrict__ dst,
                              const float* __restrict__ s1, const float* __restrict__ s2,
                              unsigned* __restrict__ rmax, int E, int N)
{
    int e = blockIdx.x * blockDim.x + threadIdx.x;
    if (e >= E) return;
    int h = blockIdx.y;
    int sr = src[e], ds = dst[e];
    atomicMax(&rmax[(size_t)h * N + sr], fkey(edge_score(s1, s2, h, N, sr, ds)));
}

__global__ void edge_attn_exp(const int* __restrict__ src, const int* __restrict__ dst,
                              const float* __restrict__ s1, const float* __restrict__ s2,
                              const unsigned* __restrict__ rmax,
                              float* __restrict__ exb, float* __restrict__ den,
                              int E, int N)
{
    int e = blockIdx.x * blockDim.x + threadIdx.x;
    if (e >= E) return;
    int h = blockIdx.y;
    int sr = src[e], ds = dst[e];
    float sc = edge_score(s1, s2, h, N, sr, ds);
    float m  = funkey(rmax[(size_t)h * N + sr]);
    float ex = __expf(sc - m);
    exb[(size_t)h * E + e] = ex;
    atomicAdd(&den[(size_t)h * N + sr], ex);
}

// out[src] += (ex/den[src]) * h[dst]; one wave per edge, 2 feats per lane.
__global__ void edge_aggregate(const int* __restrict__ src, const int* __restrict__ dst,
                               const float* __restrict__ exb, const float* __restrict__ den,
                               const float* __restrict__ Hf, float* __restrict__ Out,
                               int E, int N)
{
    const int h    = blockIdx.y;
    const int lane = threadIdx.x & 31;
    const int nw   = (gridDim.x * blockDim.x) >> 5;
    int e = (blockIdx.x * blockDim.x + threadIdx.x) >> 5;
    for (; e < E; e += nw) {
        const int sr = src[e], ds = dst[e];
        if (e + nw < E)   // pull next h[dst] row toward L2/L0 early
            __builtin_prefetch(Hf + ((size_t)h * N + dst[e + nw]) * 64, 0, 1);
        const float a = exb[(size_t)h * E + e] / den[(size_t)h * N + sr];
        const float* hv = Hf + ((size_t)h * N + ds) * 64;
        float*       ov = Out + ((size_t)h * N + sr) * 64;
        atomicAdd(&ov[lane],      a * hv[lane]);
        atomicAdd(&ov[lane + 32], a * hv[lane + 32]);
    }
}

// in: (H, N, 64) -> out: (N, H*64) with ELU (layer-1 concat=True epilogue)
__global__ void elu_concat(const float* __restrict__ in, float* __restrict__ out, int N)
{
    int idx = blockIdx.x * blockDim.x + threadIdx.x;
    if (idx >= N * NHEADS * 64) return;
    int f = idx & 63, hh = (idx >> 6) & (NHEADS - 1), node = idx >> 8;
    float v = in[((size_t)hh * N + node) * 64 + f];
    out[idx] = v > 0.f ? v : (__expf(v) - 1.0f);
}

// ---------------------------------------------------------------------------
extern "C" void kernel_launch(void* const* d_in, const int* in_sizes, int n_in,
                              void* d_out, int out_size, void* d_ws, size_t ws_size,
                              hipStream_t stream)
{
    (void)in_sizes; (void)n_in; (void)out_size; (void)ws_size;
    const float* x  = (const float*)d_in[0];
    const int*   ei = (const int*)d_in[1];     // (2, E): row0=src, row1=dst
    const float* Wh = (const float*)d_in[2];   // (H, 128, 64)
    const float* ah = (const float*)d_in[3];   // (H, 128, 1)
    const float* Wo = (const float*)d_in[4];   // (256, 64)
    const float* ao = (const float*)d_in[5];   // (128, 1)
    float* outp = (float*)d_out;               // (N, 64) f32

    const int N = NNODES, E = NEDGES, H = NHEADS;
    const int* src = ei;
    const int* dst = ei + E;

    char* w = (char*)d_ws;
    auto carve = [&](size_t bytes) -> char* {
        char* r = w; w += (bytes + 255) & ~(size_t)255; return r;
    };
    float*    h1   = (float*)   carve((size_t)H * N * 64 * 4);
    float*    s1a  = (float*)   carve((size_t)H * N * 4);
    float*    s2a  = (float*)   carve((size_t)H * N * 4);
    unsigned* rm1  = (unsigned*)carve((size_t)H * N * 4);
    float*    den1 = (float*)   carve((size_t)H * N * 4);
    float*    ex1  = (float*)   carve((size_t)H * E * 4);
    float*    out1 = (float*)   carve((size_t)H * N * 64 * 4);
    float*    hcat = (float*)   carve((size_t)N * H * 64 * 4);
    float*    h2   = (float*)   carve((size_t)N * 64 * 4);
    float*    s1b  = (float*)   carve((size_t)N * 4);
    float*    s2b  = (float*)   carve((size_t)N * 4);
    unsigned* rm2  = (unsigned*)carve((size_t)N * 4);
    float*    den2 = (float*)   carve((size_t)N * 4);
    float*    ex2  = (float*)   carve((size_t)E * 4);

    const dim3 B(256);
    // init atomically-accumulated buffers (out, denoms, row-max keys)
    fill_u32_kernel<<<(H*N*64 + 255)/256, B, 0, stream>>>((unsigned*)out1, 0u, H*N*64);
    fill_u32_kernel<<<(H*N   + 255)/256, B, 0, stream>>>(rm1,              0u, H*N);
    fill_u32_kernel<<<(H*N   + 255)/256, B, 0, stream>>>((unsigned*)den1,  0u, H*N);
    fill_u32_kernel<<<(N*64  + 255)/256, B, 0, stream>>>((unsigned*)outp,  0u, N*64);
    fill_u32_kernel<<<(N     + 255)/256, B, 0, stream>>>(rm2,              0u, N);
    fill_u32_kernel<<<(N     + 255)/256, B, 0, stream>>>((unsigned*)den2,  0u, N);

    const int gemmBlocks = (N + 255) / 256;   // 256 rows per block (2 tiles/wave)
    const int edgeBlocks = (E + 255) / 256;

    // ---- layer 1: 4 heads ----
    gat_gemm_wmma<<<dim3(gemmBlocks, H), B, 0, stream>>>(x, Wh, ah, h1, s1a, s2a, N, FIN);
    edge_attn_max<<<dim3(edgeBlocks, H), B, 0, stream>>>(src, dst, s1a, s2a, rm1, E, N);
    edge_attn_exp<<<dim3(edgeBlocks, H), B, 0, stream>>>(src, dst, s1a, s2a, rm1, ex1, den1, E, N);
    edge_aggregate<<<dim3(2048, H), B, 0, stream>>>(src, dst, ex1, den1, h1, out1, E, N);
    elu_concat<<<(N*H*64 + 255)/256, B, 0, stream>>>(out1, hcat, N);

    // ---- layer 2: single output head, K = 256 ----
    gat_gemm_wmma<<<dim3(gemmBlocks, 1), B, 0, stream>>>(hcat, Wo, ao, h2, s1b, s2b, N, H*FOUT);
    edge_attn_max<<<dim3(edgeBlocks, 1), B, 0, stream>>>(src, dst, s1b, s2b, rm2, E, N);
    edge_attn_exp<<<dim3(edgeBlocks, 1), B, 0, stream>>>(src, dst, s1b, s2b, rm2, ex2, den2, E, N);
    edge_aggregate<<<dim3(2048, 1), B, 0, stream>>>(src, dst, ex2, den2, h2, outp, E, N);
}
